// GPT_31198642438355
// MI455X (gfx1250) — matrix operand order repaired and checked by reading, not compiled, not run on hardware
//
#include <hip/hip_runtime.h>

// Problem constants (match reference)
#define B_   2
#define S_   2048
#define E_   512
#define KQ_  512
#define VD_  512
#define VOC_ 32000

typedef __attribute__((ext_vector_type(8)))  float  v8f;
typedef __attribute__((ext_vector_type(8)))  __bf16 v8bf;
typedef __attribute__((ext_vector_type(16))) __bf16 v16bf;

typedef __attribute__((ext_vector_type(4))) unsigned int u32x4;
typedef __attribute__((ext_vector_type(4))) int          i32x4;
typedef __attribute__((ext_vector_type(8))) int          i32x8;

// ---- TDM availability probes --------------------------------------------
#if defined(__has_builtin)
#if __has_builtin(__builtin_amdgcn_tensor_load_to_lds)
#define HAVE_TDM 1
#endif
#if __has_builtin(__builtin_amdgcn_s_wait_tensorcnt)
#define HAVE_WAIT_TENSOR 1
#endif
#endif
#ifndef HAVE_TDM
#define HAVE_TDM 0
#endif
#ifndef HAVE_WAIT_TENSOR
#define HAVE_WAIT_TENSOR 0
#endif
// ROCm 7.2 clang-22 declares the 5-arg builtin; therock/clang-23 (which ships
// the gfx1250 TDM header) declares a 6-arg form. Use header presence as proxy.
#if __has_include(<hip/amd_detail/amd_gfx1250_TDM.h>)
#define TDM_6ARG 1
#else
#define TDM_6ARG 0
#endif

// ---- helpers -------------------------------------------------------------

__device__ __forceinline__ unsigned short f2bf(float f) {
  unsigned int u = __float_as_uint(f);
  u += 0x7FFFu + ((u >> 16) & 1u);          // round-to-nearest-even
  return (unsigned short)(u >> 16);
}

// A-fragment (16-bit A, wave32): lane m=lane&15, half=lane>>4;
//   elems 0..15 = K[half*16..+15]
__device__ __forceinline__ v16bf load_a_frag(const unsigned short* arow, int kk) {
  v8bf lo = *(const v8bf*)(arow + kk);
  v8bf hi = *(const v8bf*)(arow + kk + 8);
  v16bf a;
#pragma unroll
  for (int i = 0; i < 8; ++i) { a[i] = lo[i]; a[i + 8] = hi[i]; }
  return a;
}

// One 16x16 (f32 acc) tile: D = A[16xK] * B^T where Bm is [N=16 rows][K] row-major.
// B-operand layout: lane col n=lane&15, half=lane>>4; elems 0..15 = K[half*16..+15]
__device__ __forceinline__ v8f wmma_tile(const unsigned short* A, int lda,
                                         const unsigned short* Bm, int ldb,
                                         int K, int lane) {
  v8f c = {0.f, 0.f, 0.f, 0.f, 0.f, 0.f, 0.f, 0.f};
  const int m    = lane & 15;
  const int half = lane >> 4;
  const unsigned short* arow = A + (long long)m * lda + half * 16;
  const unsigned short* brow = Bm + (long long)m * ldb + half * 16;
  for (int kk = 0; kk < K; kk += 32) {
    v16bf a = load_a_frag(arow, kk);
    v16bf b = *(const v16bf*)(brow + kk);
    c = __builtin_amdgcn_wmma_f32_16x16x32_bf16(false, a, false, b,
                                                (short)0, c, false, false);
  }
  return c;
}

// ---- kernels -------------------------------------------------------------

// Embedding lookup + sinusoidal PE -> h (bf16), one row per block.
__global__ void k_embed(const int* __restrict__ x, const float* __restrict__ emb,
                        unsigned short* __restrict__ hbf) {
  const int row = blockIdx.x;            // 0 .. B*S-1
  const int s   = row & (S_ - 1);
  const int tok = x[row];
  const float c0 = 9.210340371976184f / (float)E_;  // ln(10000)/d_model
  for (int e = threadIdx.x; e < E_; e += blockDim.x) {
    float div = __expf(-(float)(e & ~1) * c0);
    float ang = (float)s * div;
    float pe  = (e & 1) ? __cosf(ang) : __sinf(ang);
    float v   = emb[(long long)tok * E_ + e] + pe;
    hbf[(long long)row * E_ + e] = f2bf(v);
  }
}

__global__ void k_cvt_bf16(const float* __restrict__ src,
                           unsigned short* __restrict__ dst, long long n) {
  long long i = (long long)blockIdx.x * blockDim.x + threadIdx.x;
  if (i < n) dst[i] = f2bf(src[i]);
}

// D = h * W^T; scale folds sqrt(512) into Q; transposeV stores V^T[b][v][s].
__global__ void k_proj(const unsigned short* __restrict__ hbf,
                       const unsigned short* __restrict__ wbf,
                       unsigned short* __restrict__ out,
                       float scale, int transposeV) {
  const int lane = threadIdx.x;
  const int mt = blockIdx.x;
  const int nt = blockIdx.y * blockDim.y + threadIdx.y;
  v8f c = wmma_tile(hbf + (long long)mt * 16 * E_, E_,
                    wbf + (long long)nt * 16 * E_, E_, E_, lane);
  const int col = nt * 16 + (lane & 15);
  const int rb  = (lane >> 4) * 8;
#pragma unroll
  for (int j = 0; j < 8; ++j) {
    const int row = mt * 16 + rb + j;
    const unsigned short v = f2bf(c[j] * scale);
    if (!transposeV) {
      out[(long long)row * 512 + col] = v;
    } else {
      const int b = row / S_, sq = row & (S_ - 1);
      out[((long long)b * VD_ + col) * S_ + sq] = v;
    }
  }
}

// scores[b][q][s] for k-tile <= q-tile (causal); diagonal masked to -inf.
__global__ void k_scores(const unsigned short* __restrict__ qbf,
                         const unsigned short* __restrict__ kbf,
                         float* __restrict__ scores) {
  const int qt = blockIdx.x;
  const int kt = blockIdx.y * blockDim.y + threadIdx.y;
  const int b  = blockIdx.z;
  if (kt > qt) return;
  const int lane = threadIdx.x;
  const unsigned short* Qb = qbf + (long long)b * S_ * KQ_;
  const unsigned short* Kb = kbf + (long long)b * S_ * KQ_;
  v8f c = wmma_tile(Qb + (long long)qt * 16 * KQ_, KQ_,
                    Kb + (long long)kt * 16 * KQ_, KQ_, KQ_, lane);
  const int scol = kt * 16 + (lane & 15);
  const int rb   = (lane >> 4) * 8;
#pragma unroll
  for (int j = 0; j < 8; ++j) {
    const int qrow = qt * 16 + rb + j;
    const float v = (scol <= qrow) ? c[j] : -1e30f;
    scores[((long long)b * S_ + qrow) * S_ + scol] = v;
  }
}

// Row softmax over valid prefix; writes P (bf16), zero-fills the tail.
__global__ void k_softmax(const float* __restrict__ scores,
                          unsigned short* __restrict__ pbf) {
  const int q = blockIdx.x, b = blockIdx.y;
  const int end = ((q >> 4) + 1) << 4;
  const float* row = scores + ((long long)b * S_ + q) * S_;
  unsigned short* prow = pbf + ((long long)b * S_ + q) * S_;
  __shared__ float red[256];
  float m = -1e30f;
  for (int s = threadIdx.x; s < end; s += blockDim.x) m = fmaxf(m, row[s]);
  red[threadIdx.x] = m; __syncthreads();
  for (int o = 128; o > 0; o >>= 1) {
    if (threadIdx.x < o) red[threadIdx.x] = fmaxf(red[threadIdx.x], red[threadIdx.x + o]);
    __syncthreads();
  }
  m = red[0]; __syncthreads();
  float l = 0.f;
  for (int s = threadIdx.x; s < end; s += blockDim.x) l += __expf(row[s] - m);
  red[threadIdx.x] = l; __syncthreads();
  for (int o = 128; o > 0; o >>= 1) {
    if (threadIdx.x < o) red[threadIdx.x] += red[threadIdx.x + o];
    __syncthreads();
  }
  const float inv = 1.0f / red[0];
  for (int s = threadIdx.x; s < S_; s += blockDim.x) {
    const float p = (s < end) ? __expf(row[s] - m) * inv : 0.f;
    prow[s] = f2bf(p);
  }
}

// ctx = P * V using V^T (contiguous B-operand reads).
__global__ void k_ctx(const unsigned short* __restrict__ pbf,
                      const unsigned short* __restrict__ vtbf,
                      unsigned short* __restrict__ ctxbf) {
  const int lane = threadIdx.x;
  const int mt = blockIdx.x;
  const int nt = blockIdx.y * blockDim.y + threadIdx.y;
  const int gs0 = mt * 16;
  const int b = gs0 / S_, q0 = gs0 & (S_ - 1);
  const int kend = min(S_, (((q0 >> 4) + 1) * 16 + 31) & ~31);
  const unsigned short* Pb  = pbf  + ((long long)b * S_ + q0) * S_;
  const unsigned short* Vtb = vtbf + ((long long)b * VD_ + nt * 16) * S_;
  v8f c = wmma_tile(Pb, S_, Vtb, S_, kend, lane);
  const int col = nt * 16 + (lane & 15);
  const int rb  = (lane >> 4) * 8;
#pragma unroll
  for (int j = 0; j < 8; ++j)
    ctxbf[(long long)(gs0 + rb + j) * VD_ + col] = f2bf(c[j]);
}

// ---- logits GEMM: 128M x 64N macro-tile per block ------------------------
// 8 waves/block; Wh 64x512 bf16 slab staged to LDS (TDM if available),
// each wave: 1 M-tile x 4 N-tiles, A-frag loaded once per k-step, reused x4.
#define LOGITS_NBLK 64                       // cols per block (4 N-tiles)
#define LOGITS_MBLK 128                      // rows per block (8 M-tiles)

#if HAVE_TDM
// Issue a 2D TDM load: tile [rows x 512] bf16, row-major, into LDS offset 0.
__device__ __forceinline__ void tdm_stage_rows(const unsigned short* gsrc,
                                               unsigned int lds_off, int rows) {
  unsigned long long ga = (unsigned long long)gsrc;
  u32x4 g0;
  g0[0] = 1u;                                        // count=1, user mode
  g0[1] = lds_off;                                   // lds_addr (bytes)
  g0[2] = (unsigned int)(ga & 0xFFFFFFFFu);          // global_addr[31:0]
  g0[3] = (unsigned int)((ga >> 32) & 0x1FFFFFFu)    // global_addr[56:32]
        | (2u << 30);                                // type=2 (image)
  i32x8 g1;
  g1[0] = (int)(1u << 16);                           // data_size=2B; no mask
  g1[1] = (int)(512u << 16);                         // tensor_dim0[15:0]=512
  g1[2] = (int)((unsigned)rows << 16);               // dim0 hi=0; tensor_dim1=rows
  g1[3] = (int)(512u << 16);                         // dim1 hi=0; tile_dim0=512
  g1[4] = rows;                                      // tile_dim1=rows; tile_dim2=0
  g1[5] = 512;                                       // tensor_dim0_stride lo
  g1[6] = 0;                                         // stride0 hi, stride1 lo
  g1[7] = 0;                                         // stride1 hi
  i32x4 z4 = {0, 0, 0, 0};
#if TDM_6ARG
  i32x8 z8 = {0, 0, 0, 0, 0, 0, 0, 0};
  __builtin_amdgcn_tensor_load_to_lds(g0, g1, z4, z4, z8, 0);
#else
  __builtin_amdgcn_tensor_load_to_lds(g0, g1, z4, z4, 0);
#endif
}
#endif

__global__ void __launch_bounds__(256)
k_logits(const unsigned short* __restrict__ ctxbf,
         const unsigned short* __restrict__ whbf,
         const float* __restrict__ bh,
         float* __restrict__ out) {
  __shared__ unsigned short Bsh[LOGITS_NBLK * VD_];   // 64 KB
  const int lane = threadIdx.x;
  const int wv   = threadIdx.y;                        // 0..7
  const int mb   = blockIdx.x;                         // 0..31  (4096/128)
  const int nb   = blockIdx.y;                         // 0..499 (32000/64)
  const unsigned short* whsrc = whbf + (size_t)nb * LOGITS_NBLK * VD_;

  // ---- stage Wh slab [64 x 512] bf16 into LDS ----
#if HAVE_TDM
  if (wv == 0) {
    tdm_stage_rows(whsrc, (unsigned int)(size_t)Bsh, LOGITS_NBLK);
#if HAVE_WAIT_TENSOR
    __builtin_amdgcn_s_wait_tensorcnt(0);
#endif
  }
  __syncthreads();
#else
  {
    const uint4* src = (const uint4*)whsrc;            // 4096 x 16B
    uint4* dst = (uint4*)Bsh;
    for (int i = wv * 32 + lane; i < LOGITS_NBLK * VD_ / 8; i += 256)
      dst[i] = src[i];
    __syncthreads();
  }
#endif

  // ---- per-wave compute: 1 M-tile x 4 N-tiles ----
  const int mt = mb * 8 + wv;
  const int m    = lane & 15;
  const int half = lane >> 4;
  const unsigned short* arow = ctxbf + ((long long)mt * 16 + m) * VD_ + half * 16;

  // prefetch next block's A panel (speculative; harmless at edges)
  __builtin_prefetch(arow + 16 * VD_, 0, 3);

  v8f acc0 = {0.f,0.f,0.f,0.f,0.f,0.f,0.f,0.f};
  v8f acc1 = acc0, acc2 = acc0, acc3 = acc0;
  const unsigned short* brow = Bsh + (size_t)m * VD_ + half * 16;
  for (int kk = 0; kk < VD_; kk += 32) {
    v16bf a = load_a_frag(arow, kk);
    v16bf b0 = *(const v16bf*)(brow + kk);
    v16bf b1 = *(const v16bf*)(brow + 16 * VD_ + kk);
    v16bf b2 = *(const v16bf*)(brow + 32 * VD_ + kk);
    v16bf b3 = *(const v16bf*)(brow + 48 * VD_ + kk);
    acc0 = __builtin_amdgcn_wmma_f32_16x16x32_bf16(false, a, false, b0, (short)0, acc0, false, false);
    acc1 = __builtin_amdgcn_wmma_f32_16x16x32_bf16(false, a, false, b1, (short)0, acc1, false, false);
    acc2 = __builtin_amdgcn_wmma_f32_16x16x32_bf16(false, a, false, b2, (short)0, acc2, false, false);
    acc3 = __builtin_amdgcn_wmma_f32_16x16x32_bf16(false, a, false, b3, (short)0, acc3, false, false);
  }

  const int rb = (lane >> 4) * 8;
  const int colbase = nb * LOGITS_NBLK + (lane & 15);
  v8f* accs[4] = {&acc0, &acc1, &acc2, &acc3};
#pragma unroll
  for (int t = 0; t < 4; ++t) {
    const int col = colbase + t * 16;
    const float bias = bh[col];
    v8f c = *accs[t];
#pragma unroll
    for (int j = 0; j < 8; ++j)
      out[(long long)(mt * 16 + rb + j) * VOC_ + col] = c[j] + bias;
  }
}

// ---- launch --------------------------------------------------------------

extern "C" void kernel_launch(void* const* d_in, const int* in_sizes, int n_in,
                              void* d_out, int out_size, void* d_ws, size_t ws_size,
                              hipStream_t stream) {
  const int*   x   = (const int*)d_in[0];
  const float* emb = (const float*)d_in[1];
  const float* Wq  = (const float*)d_in[2];
  const float* Wk  = (const float*)d_in[3];
  const float* Wv  = (const float*)d_in[4];
  const float* Wh  = (const float*)d_in[5];
  const float* bh  = (const float*)d_in[6];
  float* out = (float*)d_out;

  char* ws = (char*)d_ws;
  size_t off = 0;
  auto take = [&](size_t bytes) -> char* {
    char* p = ws + off;
    off = (off + bytes + 255) & ~(size_t)255;
    return p;
  };
  // ~105 MB total workspace
  unsigned short* hbf   = (unsigned short*)take((size_t)B_ * S_ * E_ * 2);
  unsigned short* qbf   = (unsigned short*)take((size_t)B_ * S_ * KQ_ * 2);
  unsigned short* kbf   = (unsigned short*)take((size_t)B_ * S_ * KQ_ * 2);
  unsigned short* vtbf  = (unsigned short*)take((size_t)B_ * VD_ * S_ * 2);
  unsigned short* ctxbf = (unsigned short*)take((size_t)B_ * S_ * VD_ * 2);
  unsigned short* wqbf  = (unsigned short*)take((size_t)KQ_ * E_ * 2);
  unsigned short* wkbf  = (unsigned short*)take((size_t)KQ_ * E_ * 2);
  unsigned short* wvbf  = (unsigned short*)take((size_t)VD_ * E_ * 2);
  unsigned short* whbf  = (unsigned short*)take((size_t)VOC_ * VD_ * 2);
  float*          scr   = (float*)take((size_t)B_ * S_ * S_ * 4);
  unsigned short* pbf   = (unsigned short*)take((size_t)B_ * S_ * S_ * 2);

  // 1) embedding + positional encoding -> h (bf16)
  k_embed<<<B_ * S_, 256, 0, stream>>>(x, emb, hbf);

  // 2) weight f32 -> bf16 conversions
  {
    long long nw = (long long)KQ_ * E_;
    k_cvt_bf16<<<(int)((nw + 255) / 256), 256, 0, stream>>>(Wq, wqbf, nw);
    k_cvt_bf16<<<(int)((nw + 255) / 256), 256, 0, stream>>>(Wk, wkbf, nw);
    k_cvt_bf16<<<(int)((nw + 255) / 256), 256, 0, stream>>>(Wv, wvbf, nw);
    long long nh = (long long)VOC_ * VD_;
    k_cvt_bf16<<<(int)((nh + 255) / 256), 256, 0, stream>>>(Wh, whbf, nh);
  }

  // 3) Q (x sqrt(512) folded), K, V^T projections
  {
    dim3 grid(B_ * S_ / 16, KQ_ / 16 / 4), blk(32, 4);
    k_proj<<<grid, blk, 0, stream>>>(hbf, wqbf, qbf, 22.62741699796952f, 0);
    k_proj<<<grid, blk, 0, stream>>>(hbf, wkbf, kbf, 1.0f, 0);
    k_proj<<<grid, blk, 0, stream>>>(hbf, wvbf, vtbf, 1.0f, 1);
  }

  // 4) causal score tiles (f32)
  {
    dim3 grid(S_ / 16, S_ / 16 / 4, B_), blk(32, 4);
    k_scores<<<grid, blk, 0, stream>>>(qbf, kbf, scr);
  }

  // 5) softmax rows -> P (bf16, zero tail)
  k_softmax<<<dim3(S_, B_), 256, 0, stream>>>(scr, pbf);

  // 6) ctx = P V
  {
    dim3 grid(B_ * S_ / 16, VD_ / 16 / 4), blk(32, 4);
    k_ctx<<<grid, blk, 0, stream>>>(pbf, vtbf, ctxbf);
  }

  // 7) logits = ctx Wh^T + bh (fp32 output), LDS-staged Wh, 4x A-reuse
  {
    dim3 grid(B_ * S_ / LOGITS_MBLK, VOC_ / LOGITS_NBLK), blk(32, 8);
    k_logits<<<grid, blk, 0, stream>>>(ctxbf, whbf, bh, out);
  }
}
